// ProcessHandBoard_30786325577967
// MI455X (gfx1250) — compile-verified, matrix operand order checked by reading, not run on hardware
//
#include <hip/hip_runtime.h>

// ---------------------------------------------------------------------------
// Fused poker-hand MLP for MI455X (gfx1250, wave32, WMMA).
// Ref: per (b,m,combo): hero 16x32@32x128, board 16x48@48x128,
//      hidden 16x256@256x256, then out: flat(60*256) @ (15360x256).
// All GEMMs run through v_wmma_f32_16x16x32_f16 (f16 in, f32 accumulate).
// ---------------------------------------------------------------------------

typedef __attribute__((ext_vector_type(16))) _Float16 v16h;
typedef __attribute__((ext_vector_type(8)))  _Float16 v8h;
typedef __attribute__((ext_vector_type(8)))  float    v8f;

// ---- problem constants ----
constexpr int BM           = 128 * 24;       // 3072 hands
constexpr int NCOMBO       = 60;
constexpr int HANDS_PER_WG = 16;
constexpr int NWG          = BM / HANDS_PER_WG;   // 192 workgroups

// ---- f16 workspace layout (element offsets, halves) ----
constexpr long long OFF_EMB     = 0;       // 53*16 = 848
constexpr long long OFF_HANDWT  = 1024;    // 128 x 32   (n-major: Wt[n][k])
constexpr long long OFF_BOARDWT = 5120;    // 128 x 64   (K padded; rows k=48..63 zero)
constexpr long long OFF_HIDWT   = 13312;   // 256 x 256  (n-major)
constexpr long long OFF_OUTWT   = 78848;   // 60 x 256 x 256 (per combo slice, n-major)
constexpr long long CVT_TOTAL   = 848 + 4096 + 8192 + 65536 + 60LL * 65536;

// ---- LDS layout (byte offsets; all 16B aligned) ----
constexpr int S_EMB     = 0;        // 848 halves
constexpr int S_X       = 1696;     // 16*18 i32
constexpr int S_COMBO   = 2848;     // 60*5 i32
constexpr int S_BIAS    = 4048;     // 768 f32: hand(128)|board(128)|hidden(256)|out(256)
constexpr int S_AHERO   = 7120;     // 16x32 halves
constexpr int S_ABOARD  = 8144;     // 16x64 halves (cols 48..63 zero)
constexpr int S_ACT     = 10192;    // 16x256 halves (hero|board activations)
constexpr int S_HID     = 18384;    // 16x256 halves (hidden activations)
constexpr int S_HANDWT  = 26576;    // 128x32 halves
constexpr int S_BOARDWT = 34768;    // 128x64 halves
constexpr int S_HIDWT   = 51152;    // 256x256 halves (128 KB)
constexpr int S_OUTWT   = 182224;   // 256x256 halves (128 KB, per-combo slice)
constexpr int SMEM_BYTES = 313296;  // <= 320 KB per workgroup

// ---------------------------------------------------------------------------
// Kernel 0: convert f32 weights -> f16 workspace, transposed to n-major so
// WMMA B-fragments are two contiguous b128 LDS loads per lane.
// ---------------------------------------------------------------------------
__global__ void cvt_weights(const float* __restrict__ emb,
                            const float* __restrict__ handW,
                            const float* __restrict__ boardW,
                            const float* __restrict__ hidW,
                            const float* __restrict__ outW,
                            _Float16* __restrict__ ws) {
  long long i      = (long long)blockIdx.x * blockDim.x + threadIdx.x;
  long long stride = (long long)gridDim.x * blockDim.x;
  for (; i < CVT_TOTAL; i += stride) {
    long long j = i;
    if (j < 848) { ws[OFF_EMB + j] = (_Float16)emb[j]; continue; }
    j -= 848;
    if (j < 4096) {  // hand_W (32x128) -> Wt[n][k]
      int n = (int)(j / 32), k = (int)(j % 32);
      ws[OFF_HANDWT + j] = (_Float16)handW[k * 128 + n];
      continue;
    }
    j -= 4096;
    if (j < 8192) {  // board_W (48x128) -> Wt[n][k], K padded to 64
      int n = (int)(j / 64), k = (int)(j % 64);
      ws[OFF_BOARDWT + j] = (_Float16)(k < 48 ? boardW[k * 128 + n] : 0.0f);
      continue;
    }
    j -= 8192;
    if (j < 65536) {  // hidden_W (256x256) -> Wt[n][k]
      int n = (int)(j / 256), k = (int)(j % 256);
      ws[OFF_HIDWT + j] = (_Float16)hidW[k * 256 + n];
      continue;
    }
    j -= 65536;
    {  // out_W (15360x256) -> per-combo slice c, n-major: [c][n][k]
      int c = (int)(j / 65536);
      int r = (int)(j % 65536);
      int n = r / 256, k = r % 256;
      ws[OFF_OUTWT + j] = (_Float16)outW[((long long)(c * 256 + k)) * 256 + n];
    }
  }
}

// ---------------------------------------------------------------------------
// Fragment helpers (CDNA5 wave32 f16 WMMA layouts, ISA 7.12.2).
// A 16x32 f16: lane holds row (lane&15); element e -> K = e + 8*(e>>3) + 8*hi.
//   => elements 0..7 contiguous at K = kbase + 8*hi, 8..15 at K = kbase+16+8*hi.
// B 32x16 f16 (n-major storage Wt[n][k]): symmetric, lane holds column n.
// ---------------------------------------------------------------------------
__device__ __forceinline__ v16h lds_frag(const _Float16* __restrict__ base,
                                         int row, int ld, int kbase, int laneHi) {
  const _Float16* p = base + row * ld + kbase + 8 * laneHi;
  union { v16h v; v8h h[2]; } u;
  u.h[0] = *reinterpret_cast<const v8h*>(p);
  u.h[1] = *reinterpret_cast<const v8h*>(p + 16);
  return u.v;
}

__device__ __forceinline__ v8f wmma_f16(v16h a, v16h b, v8f c) {
  return __builtin_amdgcn_wmma_f32_16x16x32_f16(false, a, false, b,
                                                (short)0, c, false, false);
}

// C/D 16x16 f32: element r -> row = r + 8*hi, col = lane&15.
__device__ __forceinline__ void store_act(v8f c, _Float16* __restrict__ dst, int ld,
                                          int colBase, int lane15, int laneHi,
                                          float biasv) {
  int col   = colBase + lane15;
  int rbase = laneHi * 8;
#pragma unroll
  for (int r = 0; r < 8; ++r) {
    float v = c[r] + biasv;
    v = v > 0.0f ? v : v * 0.01f;                 // leaky_relu(0.01)
    dst[(rbase + r) * ld + col] = (_Float16)v;
  }
}

// ---------------------------------------------------------------------------
// Kernel 1: fused pipeline. 1 WG = 16 hands, 8 wave32s splitting N.
// ---------------------------------------------------------------------------
extern __shared__ char smem[];

__global__ void __launch_bounds__(256, 1)
poker_mlp_fused(const long long* __restrict__ x,     // (3072, 18) int64
                const float* __restrict__ hand_b,
                const float* __restrict__ board_b,
                const float* __restrict__ hidden_b,
                const float* __restrict__ out_b,
                const _Float16* __restrict__ ws,
                float* __restrict__ out) {           // (3072, 256) f32
  const int tid    = threadIdx.x;
  const int lane   = tid & 31;
  const int wave   = tid >> 5;
  const int lane15 = lane & 15;
  const int laneHi = lane >> 4;
  const int hand_base = blockIdx.x * HANDS_PER_WG;

  _Float16* embL     = (_Float16*)(smem + S_EMB);
  int*      xL       = (int*)     (smem + S_X);
  int*      comboL   = (int*)     (smem + S_COMBO);
  float*    biasL    = (float*)   (smem + S_BIAS);
  _Float16* aheroL   = (_Float16*)(smem + S_AHERO);
  _Float16* aboardL  = (_Float16*)(smem + S_ABOARD);
  _Float16* actL     = (_Float16*)(smem + S_ACT);
  _Float16* hidL     = (_Float16*)(smem + S_HID);
  _Float16* handWtL  = (_Float16*)(smem + S_HANDWT);
  _Float16* boardWtL = (_Float16*)(smem + S_BOARDWT);
  _Float16* hidWtL   = (_Float16*)(smem + S_HIDWT);
  _Float16* outWtL   = (_Float16*)(smem + S_OUTWT);

  // ---------------- one-time staging ----------------
  for (int i = tid; i < 848; i += 256) embL[i] = ws[OFF_EMB + i];
  {
    const uint4* s = (const uint4*)(ws + OFF_HANDWT);
    uint4* d = (uint4*)handWtL;
    for (int i = tid; i < 4096 / 8; i += 256) d[i] = s[i];
  }
  {
    const uint4* s = (const uint4*)(ws + OFF_BOARDWT);
    uint4* d = (uint4*)boardWtL;
    for (int i = tid; i < 8192 / 8; i += 256) d[i] = s[i];
  }
  {
    const uint4* s = (const uint4*)(ws + OFF_HIDWT);
    uint4* d = (uint4*)hidWtL;
    for (int i = tid; i < 65536 / 8; i += 256) d[i] = s[i];
  }
  for (int i = tid; i < 16 * 18; i += 256)
    xL[i] = (int)x[(long long)(hand_base + i / 18) * 18 + (i % 18)];
  for (int i = tid; i < 128; i += 256) biasL[i]       = hand_b[i];
  for (int i = tid; i < 128; i += 256) biasL[128 + i] = board_b[i];
  for (int i = tid; i < 256; i += 256) biasL[256 + i] = hidden_b[i];
  for (int i = tid; i < 256; i += 256) biasL[512 + i] = out_b[i];
  // zero pad of A_board (K = 48..63) — never overwritten later
  for (int i = tid; i < 16 * 16; i += 256)
    aboardL[(i / 16) * 64 + 48 + (i % 16)] = (_Float16)0.0f;
  if (tid == 0) {  // combo table, reference order: hero pairs x board triples
    int ci = 0;
    for (int i = 0; i < 4; ++i)
      for (int j = i + 1; j < 4; ++j)
        for (int a = 0; a < 5; ++a)
          for (int b = a + 1; b < 5; ++b)
            for (int c = b + 1; c < 5; ++c) {
              comboL[ci * 5 + 0] = i;
              comboL[ci * 5 + 1] = j;
              comboL[ci * 5 + 2] = 4 + a;
              comboL[ci * 5 + 3] = 4 + b;
              comboL[ci * 5 + 4] = 4 + c;
              ++ci;
            }
  }
  __syncthreads();

  const _Float16* gOutWt = ws + OFF_OUTWT;
  v8f acc0 = {};   // persistent out-layer accumulators (two 16-col tiles / wave)
  v8f acc1 = {};

  for (int c = 0; c < NCOMBO; ++c) {
    __syncthreads();  // previous iteration's readers of A/outWt are done

    // ---- stage 1: embedding gather + stage this combo's out_W slice ----
    for (int idx = tid; idx < 16 * 5 * 16; idx += 256) {
      int hh  = idx / 80;
      int rem = idx % 80;
      int s   = rem / 16;
      int e   = rem % 16;
      int pos  = comboL[c * 5 + s];
      int rank = xL[hh * 18 + 2 * pos];
      int suit = xL[hh * 18 + 2 * pos + 1];
      int card = (rank > 0 ? rank - 1 : 0) + (suit > 0 ? (suit - 1) * 13 : 0);
      _Float16 v = embL[card * 16 + e];
      if (s < 2) aheroL[hh * 32 + s * 16 + e] = v;
      else       aboardL[hh * 64 + (s - 2) * 16 + e] = v;
    }
    {
      const uint4* s = (const uint4*)(gOutWt + (long long)c * 65536);
      uint4* d = (uint4*)outWtL;
      for (int i = tid; i < 8192; i += 256) d[i] = s[i];
      if (c + 1 < NCOMBO)  // global_prefetch_b8 next slice toward L2/L0
        __builtin_prefetch(gOutWt + (long long)(c + 1) * 65536 + tid * 32, 0, 0);
    }
    __syncthreads();

    // ---- stage 2: hero (K=32) + board (K=64 padded), N=128 over 8 waves ----
    {
      int n0 = wave * 16;
      v16h a  = lds_frag(aheroL, lane15, 32, 0, laneHi);
      v16h b  = lds_frag(handWtL, n0 + lane15, 32, 0, laneHi);
      v8f  ch = {};
      ch = wmma_f16(a, b, ch);
      store_act(ch, actL, 256, n0, lane15, laneHi, biasL[n0 + lane15]);

      v8f cb = {};
#pragma unroll
      for (int kk = 0; kk < 2; ++kk) {
        v16h ab = lds_frag(aboardL, lane15, 64, kk * 32, laneHi);
        v16h bb = lds_frag(boardWtL, n0 + lane15, 64, kk * 32, laneHi);
        cb = wmma_f16(ab, bb, cb);
      }
      store_act(cb, actL, 256, 128 + n0, lane15, laneHi, biasL[128 + n0 + lane15]);
    }
    __syncthreads();

    // ---- stage 3: hidden 16x256 @ 256x256; each wave: 2 N-tiles x 8 K-steps ----
#pragma unroll
    for (int t = 0; t < 2; ++t) {
      int n0 = wave * 16 + t * 128;
      v8f ch = {};
#pragma unroll
      for (int kk = 0; kk < 8; ++kk) {
        v16h a = lds_frag(actL, lane15, 256, kk * 32, laneHi);
        v16h b = lds_frag(hidWtL, n0 + lane15, 256, kk * 32, laneHi);
        ch = wmma_f16(a, b, ch);
      }
      store_act(ch, hidL, 256, n0, lane15, laneHi, biasL[256 + n0 + lane15]);
    }
    __syncthreads();

    // ---- stage 4: out-layer accumulation into persistent registers ----
#pragma unroll
    for (int t = 0; t < 2; ++t) {
      int n0 = wave * 16 + t * 128;
      v8f a_acc = t ? acc1 : acc0;
#pragma unroll
      for (int kk = 0; kk < 8; ++kk) {
        v16h a = lds_frag(hidL, lane15, 256, kk * 32, laneHi);
        v16h b = lds_frag(outWtL, n0 + lane15, 256, kk * 32, laneHi);
        a_acc = wmma_f16(a, b, a_acc);
      }
      if (t) acc1 = a_acc; else acc0 = a_acc;
    }
  }

  // ---------------- epilogue: bias + leaky relu + store f32 ----------------
#pragma unroll
  for (int t = 0; t < 2; ++t) {
    int n0  = wave * 16 + t * 128;
    int col = n0 + lane15;
    float b = biasL[512 + col];
    v8f acc = t ? acc1 : acc0;
#pragma unroll
    for (int r = 0; r < 8; ++r) {
      float v = acc[r] + b;
      v = v > 0.0f ? v : v * 0.01f;
      out[(long long)(hand_base + laneHi * 8 + r) * 256 + col] = v;
    }
  }
}

// ---------------------------------------------------------------------------
extern "C" void kernel_launch(void* const* d_in, const int* in_sizes, int n_in,
                              void* d_out, int out_size, void* d_ws, size_t ws_size,
                              hipStream_t stream) {
  const long long* x        = (const long long*)d_in[0];
  const float*     card_emb = (const float*)d_in[1];
  const float*     hand_W   = (const float*)d_in[2];
  const float*     hand_b   = (const float*)d_in[3];
  const float*     board_W  = (const float*)d_in[4];
  const float*     board_b  = (const float*)d_in[5];
  const float*     hidden_W = (const float*)d_in[6];
  const float*     hidden_b = (const float*)d_in[7];
  const float*     out_W    = (const float*)d_in[8];
  const float*     out_b    = (const float*)d_in[9];
  _Float16* ws16 = (_Float16*)d_ws;
  float*    out  = (float*)d_out;

  int cvt_blocks = (int)((CVT_TOTAL + 255) / 256);
  cvt_weights<<<cvt_blocks, 256, 0, stream>>>(card_emb, hand_W, board_W,
                                              hidden_W, out_W, ws16);

  (void)hipFuncSetAttribute((const void*)poker_mlp_fused,
                            hipFuncAttributeMaxDynamicSharedMemorySize,
                            SMEM_BYTES);
  poker_mlp_fused<<<NWG, 256, SMEM_BYTES, stream>>>(x, hand_b, board_b,
                                                    hidden_b, out_b, ws16, out);
}
